// Linear_Attention_996432413015
// MI455X (gfx1250) — matrix operand order repaired and checked by reading, not compiled, not run on hardware
//
#include <hip/hip_runtime.h>
#include <hip/hip_bf16.h>

// ---------------------------------------------------------------------------
// R=16, C=2048, E=256, H=8, hd=32 -> m=c*8+h (16384 rows), d=r*32+i (512).
// All GEMMs: f16 WMMA 16x16x32, f32 accum, 2x4-tile (32x64) register blocking.
// ---------------------------------------------------------------------------

typedef _Float16 h8v  __attribute__((ext_vector_type(8)));
typedef _Float16 h16v __attribute__((ext_vector_type(16)));
typedef float    f8v  __attribute__((ext_vector_type(8)));

union FragU { h16v v; h8v h[2]; };

// A-fragment (16x32 f16), CDNA5 dense-A layout:
// lanes 0-15: VGPR0-3 K=0..7, VGPR4-7 K=16..23 ; lanes 16-31: K ranges +8.
__device__ __forceinline__ h16v load_frag_a(const _Float16* __restrict__ base,
                                            int row, int kbase, int stride) {
  const int lane = threadIdx.x & 31;
  const int r  = row + (lane & 15);
  const int hf = lane >> 4;
  const _Float16* p = base + (size_t)r * stride + kbase + hf * 8;
  FragU f;
  f.h[0] = *(const h8v*)(p);        // K = kbase + hf*8 + 0..7
  f.h[1] = *(const h8v*)(p + 16);   // K = kbase + 16 + hf*8 + 0..7
  return f.v;
}

// B-fragment (32x16 f16), same K order as A (lanes 0-15: K=0..7,16..23 for N=lane,
// lanes 16-31: K=8..15,24..31). Source stored as [n][k] row-major (B^T),
// contraction contiguous -> two 16B chunks per lane.
__device__ __forceinline__ h16v load_frag_b(const _Float16* __restrict__ base,
                                            int col, int kbase, int stride) {
  const int lane = threadIdx.x & 31;
  const int n  = col + (lane & 15);
  const int hf = lane >> 4;
  const _Float16* p = base + (size_t)n * stride + kbase + hf * 8;
  FragU f;
  f.h[0] = *(const h8v*)(p);
  f.h[1] = *(const h8v*)(p + 16);
  return f.v;
}

__device__ __forceinline__ f8v wmma16(h16v a, h16v b, f8v c) {
  return __builtin_amdgcn_wmma_f32_16x16x32_f16(false, a, false, b,
                                                (short)0, c, false, false);
}

// 32x64 register-blocked GEMM core: 2 A-frags + 4 B-frags -> 8 WMMAs per
// k-slice (21.3 FLOP per cache byte vs 8 for unblocked).
template <int KTILES>
__device__ __forceinline__ void gemm_2x4(const _Float16* __restrict__ A,
                                         const _Float16* __restrict__ B,
                                         int m0, int n0, int sa, int sb,
                                         f8v acc[2][4]) {
#pragma unroll 2
  for (int kt = 0; kt < KTILES; ++kt) {
    const int kb = kt * 32;
    h16v a0 = load_frag_a(A, m0,      kb, sa);
    h16v a1 = load_frag_a(A, m0 + 16, kb, sa);
    h16v b0 = load_frag_b(B, n0,      kb, sb);
    h16v b1 = load_frag_b(B, n0 + 16, kb, sb);
    h16v b2 = load_frag_b(B, n0 + 32, kb, sb);
    h16v b3 = load_frag_b(B, n0 + 48, kb, sb);
    acc[0][0] = wmma16(a0, b0, acc[0][0]);
    acc[0][1] = wmma16(a0, b1, acc[0][1]);
    acc[0][2] = wmma16(a0, b2, acc[0][2]);
    acc[0][3] = wmma16(a0, b3, acc[0][3]);
    acc[1][0] = wmma16(a1, b0, acc[1][0]);
    acc[1][1] = wmma16(a1, b1, acc[1][1]);
    acc[1][2] = wmma16(a1, b2, acc[1][2]);
    acc[1][3] = wmma16(a1, b3, acc[1][3]);
  }
}

// ---------------------------------------------------------------------------
// K0: fp32 -> f16 conversion
// ---------------------------------------------------------------------------
__global__ void f32_to_f16_kernel(const float* __restrict__ src,
                                  _Float16* __restrict__ dst, int n) {
  int i = blockIdx.x * blockDim.x + threadIdx.x;
  if (i < n) dst[i] = (_Float16)src[i];
}

// ---------------------------------------------------------------------------
// K1: Y = Xh[32768,256] @ Wh^T + b, scattered into heads layout.
// mode 0: f32 -> P[m=c*8+h][d=r*32+i] ([16384,512])
// mode 1: f16 -> VhT[h'][d][c'] ([8][512][2048])
// Wave computes 32x64; 1024 x 4 wave-grid; K = 256/32 = 8 slices.
// ---------------------------------------------------------------------------
__global__ void __launch_bounds__(128)
qkv_gemm_kernel(const _Float16* __restrict__ Xh,
                const _Float16* __restrict__ Wh,
                const float* __restrict__ bias,
                float* __restrict__ outF,
                _Float16* __restrict__ vT,
                int mode) {
  const int wave = threadIdx.x >> 5;
  const int lane = threadIdx.x & 31;
  const int tid  = blockIdx.x * 4 + wave;
  const int m0 = (tid >> 2) * 32;   // row block (rc)
  const int n0 = (tid & 3) * 64;    // col block (e_out)

  f8v acc[2][4] = {};
  gemm_2x4<8>(Xh, Wh, m0, n0, 256, 256, acc);

  const int hf = lane >> 4;
#pragma unroll
  for (int mi = 0; mi < 2; ++mi) {
#pragma unroll
    for (int ni = 0; ni < 4; ++ni) {
      const int n    = n0 + ni * 16 + (lane & 15);
      const float bn = bias[n];
      const int hsel = n >> 5, ii = n & 31;
#pragma unroll
      for (int j = 0; j < 8; ++j) {
        const int rc = m0 + mi * 16 + j + 8 * hf;
        const int r = rc >> 11, c = rc & 2047;
        const float v = acc[mi][ni][j] + bn;
        const int m2 = c * 8 + hsel;      // 0..16383
        const int d  = r * 32 + ii;       // 0..511
        if (mode == 0) {
          outF[(size_t)m2 * 512 + d] = v;
        } else {
          const int hp = m2 >> 11, cp = m2 & 2047;
          vT[((size_t)(hp * 512 + d)) * 2048 + cp] = (_Float16)v;
        }
      }
    }
  }
}

// ---------------------------------------------------------------------------
// K2a: q row-softmax over d (512), scale by sqrt(512), write f16 Qh[16384,512]
// ---------------------------------------------------------------------------
__global__ void q_softmax_kernel(const float* __restrict__ Q2,
                                 _Float16* __restrict__ Qh) {
  __shared__ float red[128];
  const int row = blockIdx.x;
  const int t   = threadIdx.x;
  const float* src = Q2 + (size_t)row * 512;
  float x0 = src[t], x1 = src[t + 128], x2 = src[t + 256], x3 = src[t + 384];
  float m = fmaxf(fmaxf(x0, x1), fmaxf(x2, x3));
  red[t] = m; __syncthreads();
#pragma unroll
  for (int s = 64; s > 0; s >>= 1) {
    if (t < s) red[t] = fmaxf(red[t], red[t + s]);
    __syncthreads();
  }
  m = red[0];
  __syncthreads();
  float e0 = __expf(x0 - m), e1 = __expf(x1 - m);
  float e2 = __expf(x2 - m), e3 = __expf(x3 - m);
  red[t] = e0 + e1 + e2 + e3; __syncthreads();
#pragma unroll
  for (int s = 64; s > 0; s >>= 1) {
    if (t < s) red[t] += red[t + s];
    __syncthreads();
  }
  const float scale = 22.62741699796952f / red[0];  // sqrt(512)/sum
  _Float16* dst = Qh + (size_t)row * 512;
  dst[t]       = (_Float16)(e0 * scale);
  dst[t + 128] = (_Float16)(e1 * scale);
  dst[t + 256] = (_Float16)(e2 * scale);
  dst[t + 384] = (_Float16)(e3 * scale);
}

// ---------------------------------------------------------------------------
// K2b: k column-softmax over c' (2048) per (head,d); writes Kh[16384,512] f16
// and KhT[8][512][2048] f16. One block per (head,d); 256 threads x 8 rows.
// ---------------------------------------------------------------------------
__global__ void k_softmax_kernel(const float* __restrict__ K2,
                                 _Float16* __restrict__ Kh,
                                 _Float16* __restrict__ KhT) {
  __shared__ float red[256];
  const int col = blockIdx.x & 511;
  const int hh  = blockIdx.x >> 9;
  const int t   = threadIdx.x;
  float x[8];
#pragma unroll
  for (int k = 0; k < 8; ++k)
    x[k] = K2[((size_t)(hh * 2048 + t + k * 256)) * 512 + col];
  float m = x[0];
#pragma unroll
  for (int k = 1; k < 8; ++k) m = fmaxf(m, x[k]);
  red[t] = m; __syncthreads();
#pragma unroll
  for (int s = 128; s > 0; s >>= 1) {
    if (t < s) red[t] = fmaxf(red[t], red[t + s]);
    __syncthreads();
  }
  m = red[0];
  __syncthreads();
  float e[8]; float sum = 0.f;
#pragma unroll
  for (int k = 0; k < 8; ++k) { e[k] = __expf(x[k] - m); sum += e[k]; }
  red[t] = sum; __syncthreads();
#pragma unroll
  for (int s = 128; s > 0; s >>= 1) {
    if (t < s) red[t] += red[t + s];
    __syncthreads();
  }
  const float inv = 1.0f / red[0];
#pragma unroll
  for (int k = 0; k < 8; ++k) {
    const int nrow = t + k * 256;
    const _Float16 val = (_Float16)(e[k] * inv);
    Kh[((size_t)(hh * 2048 + nrow)) * 512 + col] = val;
    KhT[((size_t)(hh * 512 + col)) * 2048 + nrow] = val;
  }
}

// ---------------------------------------------------------------------------
// K3: attn[h] = Q[h] @ K[h]^T -> [2048,2048] f32. Wave: 32x64 region;
// 64 x 32 wave-grid per head; K = 512/32 = 16 slices.
// ---------------------------------------------------------------------------
__global__ void __launch_bounds__(128)
attn_kernel(const _Float16* __restrict__ Qh,
            const _Float16* __restrict__ Kh,
            float* __restrict__ attn) {
  const int wave = threadIdx.x >> 5;
  const int lane = threadIdx.x & 31;
  const int tid  = blockIdx.x * 4 + wave;
  const int h    = tid >> 11;
  const int rem  = tid & 2047;
  const int m0 = (rem >> 5) * 32;   // q rows
  const int n0 = (rem & 31) * 64;   // k rows (attn cols)
  const _Float16* Qb = Qh + (size_t)h * 2048 * 512;
  const _Float16* Kb = Kh + (size_t)h * 2048 * 512;
  __builtin_prefetch(Qb + (size_t)m0 * 512, 0, 3);   // global_prefetch_b8
  __builtin_prefetch(Kb + (size_t)n0 * 512, 0, 3);

  f8v acc[2][4] = {};
  gemm_2x4<16>(Qb, Kb, m0, n0, 512, 512, acc);

  float* ob = attn + (size_t)h * 2048 * 2048;
  const int N = lane & 15, hf = lane >> 4;
#pragma unroll
  for (int mi = 0; mi < 2; ++mi)
#pragma unroll
    for (int ni = 0; ni < 4; ++ni)
#pragma unroll
      for (int j = 0; j < 8; ++j)
        ob[(size_t)(m0 + mi * 16 + j + 8 * hf) * 2048 + n0 + ni * 16 + N] =
            acc[mi][ni][j];
}

// ---------------------------------------------------------------------------
// K4: ctx[h] = K[h]^T @ V[h] (contract n=2048), A=KhT rows, B=VhT rows.
// Stores ctx transposed f16: ctxT[h][dd][d]. Wave: 32x64; 16 x 8 grid/head;
// K = 2048/32 = 64 slices.
// ---------------------------------------------------------------------------
__global__ void __launch_bounds__(128)
ctx_kernel(const _Float16* __restrict__ KhT,
           const _Float16* __restrict__ VhT,
           _Float16* __restrict__ ctxT) {
  const int wave = threadIdx.x >> 5;
  const int lane = threadIdx.x & 31;
  const int tid  = blockIdx.x * 4 + wave;
  const int h    = tid >> 7;
  const int rem  = tid & 127;
  const int m0 = (rem >> 3) * 32;   // d block
  const int n0 = (rem & 7) * 64;    // dd block
  const _Float16* Ab = KhT + (size_t)h * 512 * 2048;
  const _Float16* Bb = VhT + (size_t)h * 512 * 2048;
  __builtin_prefetch(Ab + (size_t)m0 * 2048, 0, 3);
  __builtin_prefetch(Bb + (size_t)n0 * 2048, 0, 3);

  f8v acc[2][4] = {};
  gemm_2x4<64>(Ab, Bb, m0, n0, 2048, 2048, acc);

  const int N = lane & 15, hf = lane >> 4;
#pragma unroll
  for (int mi = 0; mi < 2; ++mi)
#pragma unroll
    for (int ni = 0; ni < 4; ++ni)
#pragma unroll
      for (int j = 0; j < 8; ++j) {
        const int d  = m0 + mi * 16 + j + 8 * hf;
        const int dd = n0 + ni * 16 + N;
        ctxT[((size_t)(h * 512 + dd)) * 512 + d] = (_Float16)acc[mi][ni][j];
      }
}

// ---------------------------------------------------------------------------
// K5: out_pre[h] = Q[h] @ ctx[h] ([2048,512] x [512,512]); scatter to final
// layout out[r][c'][h*32+i] with dd = r*32+i. Wave: 32x64; 64 x 8 grid/head;
// K = 512/32 = 16 slices.
// ---------------------------------------------------------------------------
__global__ void __launch_bounds__(128)
out_kernel(const _Float16* __restrict__ Qh,
           const _Float16* __restrict__ ctxT,
           float* __restrict__ out) {
  const int wave = threadIdx.x >> 5;
  const int lane = threadIdx.x & 31;
  const int tid  = blockIdx.x * 4 + wave;
  const int h    = tid >> 9;
  const int rem  = tid & 511;
  const int m0 = (rem >> 3) * 32;   // c' block
  const int n0 = (rem & 7) * 64;    // dd block
  const _Float16* Ab = Qh + (size_t)h * 2048 * 512;
  const _Float16* Bb = ctxT + (size_t)h * 512 * 512;
  __builtin_prefetch(Ab + (size_t)m0 * 512, 0, 3);
  __builtin_prefetch(Bb + (size_t)n0 * 512, 0, 3);

  f8v acc[2][4] = {};
  gemm_2x4<16>(Ab, Bb, m0, n0, 512, 512, acc);

  const int N = lane & 15, hf = lane >> 4;
#pragma unroll
  for (int mi = 0; mi < 2; ++mi)
#pragma unroll
    for (int ni = 0; ni < 4; ++ni) {
      const int dd = n0 + ni * 16 + N;
      const int r = dd >> 5, ii = dd & 31;
#pragma unroll
      for (int j = 0; j < 8; ++j) {
        const int cp = m0 + mi * 16 + j + 8 * hf;
        out[((size_t)(r * 2048 + cp)) * 256 + h * 32 + ii] = acc[mi][ni][j];
      }
    }
}

// ---------------------------------------------------------------------------
// Launch
// ---------------------------------------------------------------------------
extern "C" void kernel_launch(void* const* d_in, const int* in_sizes, int n_in,
                              void* d_out, int out_size, void* d_ws, size_t ws_size,
                              hipStream_t stream) {
  const float* x  = (const float*)d_in[0];
  const float* Wq = (const float*)d_in[1];
  const float* bq = (const float*)d_in[2];
  const float* Wk = (const float*)d_in[3];
  const float* bk = (const float*)d_in[4];
  const float* Wv = (const float*)d_in[5];
  const float* bv = (const float*)d_in[6];

  float* out  = (float*)d_out;                    // [16,2048,256]  = 8388608
  float* attn = (float*)d_out + 8388608;          // [8,2048,2048]  = 33554432

  char* ws = (char*)d_ws;
  size_t off = 0;
  auto carve = [&](size_t bytes) -> void* {
    void* p = ws + off;
    off = (off + bytes + 255) & ~(size_t)255;
    return p;
  };
  _Float16* Xh   = (_Float16*)carve(32768ull * 256 * 2);   // x as f16 [32768,256]
  _Float16* Whq  = (_Float16*)carve(256ull * 256 * 2);
  _Float16* Whk  = (_Float16*)carve(256ull * 256 * 2);
  _Float16* Whv  = (_Float16*)carve(256ull * 256 * 2);
  float*    Q2   = (float*)carve(16384ull * 512 * 4);      // q pre-softmax
  float*    K2   = (float*)carve(16384ull * 512 * 4);      // k pre-softmax
  _Float16* Qh   = (_Float16*)carve(16384ull * 512 * 2);   // softmax(q)*sqrt(d)
  _Float16* Kh   = (_Float16*)carve(16384ull * 512 * 2);   // softmax(k)
  _Float16* KhT  = (_Float16*)carve(8ull * 512 * 2048 * 2);
  _Float16* VhT  = (_Float16*)carve(8ull * 512 * 2048 * 2);
  _Float16* ctxT = (_Float16*)carve(8ull * 512 * 512 * 2);

  // convert inputs to f16
  f32_to_f16_kernel<<<32768, 256, 0, stream>>>(x,  Xh,  8388608);
  f32_to_f16_kernel<<<256,   256, 0, stream>>>(Wq, Whq, 65536);
  f32_to_f16_kernel<<<256,   256, 0, stream>>>(Wk, Whk, 65536);
  f32_to_f16_kernel<<<256,   256, 0, stream>>>(Wv, Whv, 65536);

  // projections (WMMA) with heads-layout scatter: 4096 waves each
  qkv_gemm_kernel<<<1024, 128, 0, stream>>>(Xh, Whq, bq, Q2, nullptr, 0);
  qkv_gemm_kernel<<<1024, 128, 0, stream>>>(Xh, Whk, bk, K2, nullptr, 0);
  qkv_gemm_kernel<<<1024, 128, 0, stream>>>(Xh, Whv, bv, nullptr, VhT, 1);

  // softmaxes
  q_softmax_kernel<<<16384, 128, 0, stream>>>(Q2, Qh);
  k_softmax_kernel<<<4096,  256, 0, stream>>>(K2, Kh, KhT);

  // attention weights (output #2): 16384 waves
  attn_kernel<<<4096, 128, 0, stream>>>(Qh, Kh, attn);

  // context (1024 waves) and final output (4096 waves)
  ctx_kernel<<<256,  128, 0, stream>>>(KhT, VhT, ctxT);
  out_kernel<<<1024, 128, 0, stream>>>(Qh, ctxT, out);
}
